// SlidingWindowAttention_42932493091245
// MI455X (gfx1250) — matrix operand order repaired and checked by reading, not compiled, not run on hardware
//
#include <hip/hip_runtime.h>
#include <hip/hip_bf16.h>

#define HIDDEN 768
#define NHEADS 12
#define HDIM   64
#define SEQ    2048
#define WIN    128
#define NEGV   (-1.0e30f)

typedef __bf16 v16bf __attribute__((ext_vector_type(16)));
typedef float  v8f   __attribute__((ext_vector_type(8)));

// ---------------------------------------------------------------------------
// optional CDNA5 async global->LDS copy (ASYNCcnt path), guarded so the file
// compiles on toolchains that do not expose the builtin.
// Probe-learned signature: (int4 addrspace(1)* gsrc, int4 addrspace(3)* ldst,
//                           imm int offset, imm int cpol)
// ---------------------------------------------------------------------------
#if __has_builtin(__builtin_amdgcn_global_load_async_to_lds_b128)
#define HAVE_ASYNC_LDS 1
typedef int v4i_vs __attribute__((vector_size(4 * sizeof(int))));
typedef __attribute__((address_space(1))) v4i_vs g_v4i;
typedef __attribute__((address_space(3))) v4i_vs l_v4i;
#define ASYNC_B128(gsrc, ldst, off)                                          \
  __builtin_amdgcn_global_load_async_to_lds_b128((g_v4i*)(gsrc),             \
                                                 (l_v4i*)(ldst), (off), 0)
#if __has_builtin(__builtin_amdgcn_s_wait_asynccnt)
#define ASYNC_WAIT() __builtin_amdgcn_s_wait_asynccnt(0)
#else
#define ASYNC_WAIT() asm volatile("s_wait_asynccnt 0" ::: "memory")
#endif
#else
#define HAVE_ASYNC_LDS 0
#define ASYNC_WAIT()
#endif

// ---------------------------------------------------------------------------
// WMMA helper: D = A(16x32 bf16) * B(32x16 bf16) + C (f32)
// ---------------------------------------------------------------------------
__device__ __forceinline__ v8f wmma_bf16(v16bf a, v16bf b, v8f c) {
  return __builtin_amdgcn_wmma_f32_16x16x32_bf16(
      /*neg_a=*/false, a, /*neg_b=*/false, b,
      /*c_mod=*/(short)0, c, /*reuse_a=*/false, /*reuse_b=*/false);
}

// A-fragment (16x32, 16-bit): lane holds row (lane&15); half = lane>>4.
// K = 16*half + e (16 consecutive)
// caller passes p = row_ptr + kbase + 16*half
__device__ __forceinline__ v16bf ldA(const __bf16* p) {
  v16bf r;
#pragma unroll
  for (int i = 0; i < 16; ++i) r[i] = p[i];
  return r;
}
// B-fragment (32x16, 16-bit): lane holds col (lane&15); K = 16*half + e
// (16 consecutive). caller passes p = col_ptr + kbase + 16*half
__device__ __forceinline__ v16bf ldB(const __bf16* p) {
  v16bf r;
#pragma unroll
  for (int i = 0; i < 16; ++i) r[i] = p[i];
  return r;
}

// cross-lane max inside each 16-lane half-group via ds_swizzle (xor mode:
// and_mask=0x1f, or_mask=0, xor_mask=m; m<16 never crosses the half boundary)
#define SWZ_MAX_STEP(v, imm)                                                  \
  v = fmaxf(v, __uint_as_float(                                               \
             __builtin_amdgcn_ds_swizzle((int)__float_as_uint(v), (imm))))
__device__ __forceinline__ float rowmax16(float v) {
  SWZ_MAX_STEP(v, 0x7C01);
  SWZ_MAX_STEP(v, 0x7C02);
  SWZ_MAX_STEP(v, 0x7C04);
  SWZ_MAX_STEP(v, 0x7C08);
  return v;
}

// ---------------------------------------------------------------------------
// fp32 -> bf16 cast (one-time; all GEMMs then run on the matrix pipe)
// ---------------------------------------------------------------------------
__global__ void cvt_f32_to_bf16(const float* __restrict__ in,
                                __bf16* __restrict__ out, int n) {
  int i = blockIdx.x * blockDim.x + threadIdx.x;
  if (i < n) out[i] = (__bf16)in[i];
}

// ---------------------------------------------------------------------------
// Y[M x N] = A[M x K](bf16) * W[N x K]^T(bf16) + bias (f32 accum)
// one wave computes a 32x32 tile; 8 waves / block (waves in a block share the
// same W tile -> WGP$ hits on B fragments).
// ---------------------------------------------------------------------------
template <bool BF16OUT>
__global__ void __launch_bounds__(256) gemm_bias_wmma(
    const __bf16* __restrict__ A, const __bf16* __restrict__ W,
    const float* __restrict__ bias, void* __restrict__ outv,
    int M, int N, int K) {
  const int lane = threadIdx.x & 31;
  const int wid  = threadIdx.x >> 5;
  const int half = lane >> 4;
  const int l16  = lane & 15;
  const int tilesM = M >> 5, tilesN = N >> 5;
  const int tid = blockIdx.x * (blockDim.x >> 5) + wid;
  if (tid >= tilesM * tilesN) return;
  const int m0 = (tid % tilesM) << 5;
  const int n0 = (tid / tilesM) << 5;

  const __bf16* a0p = A + (size_t)(m0 + l16) * K + 16 * half;
  const __bf16* a1p = a0p + (size_t)16 * K;
  const __bf16* b0p = W + (size_t)(n0 + l16) * K + 16 * half;
  const __bf16* b1p = b0p + (size_t)16 * K;

  v8f c00 = {}, c01 = {}, c10 = {}, c11 = {};
  for (int k = 0; k < K; k += 32) {
    v16bf a0 = ldA(a0p + k);
    v16bf a1 = ldA(a1p + k);
    v16bf b0 = ldB(b0p + k);
    v16bf b1 = ldB(b1p + k);
    if (k + 32 < K) {                      // global_prefetch_b8 next k-slab
      __builtin_prefetch(a0p + k + 32, 0, 0);
      __builtin_prefetch(b0p + k + 32, 0, 0);
    }
    c00 = wmma_bf16(a0, b0, c00);
    c01 = wmma_bf16(a0, b1, c01);
    c10 = wmma_bf16(a1, b0, c10);
    c11 = wmma_bf16(a1, b1, c11);
  }

  const float bi0 = bias[n0 + l16];
  const float bi1 = bias[n0 + 16 + l16];
#pragma unroll
  for (int r = 0; r < 8; ++r) {
    const int row0 = m0 + r + 8 * half, row1 = row0 + 16;
    const int col0 = n0 + l16,          col1 = col0 + 16;
    const float v00 = c00[r] + bi0, v01 = c01[r] + bi1;
    const float v10 = c10[r] + bi0, v11 = c11[r] + bi1;
    if (BF16OUT) {
      __bf16* o = (__bf16*)outv;
      o[(size_t)row0 * N + col0] = (__bf16)v00;
      o[(size_t)row0 * N + col1] = (__bf16)v01;
      o[(size_t)row1 * N + col0] = (__bf16)v10;
      o[(size_t)row1 * N + col1] = (__bf16)v11;
    } else {
      float* o = (float*)outv;
      o[(size_t)row0 * N + col0] = v00;
      o[(size_t)row0 * N + col1] = v01;
      o[(size_t)row1 * N + col0] = v10;
      o[(size_t)row1 * N + col1] = v11;
    }
  }
}

// ---------------------------------------------------------------------------
// Flash attention with window|global mask.
// block = 128 threads (4 waves) -> 64 query rows of one (b,h).
// K chunk: async global->LDS (ASYNCcnt). V chunk: register transpose to LDS.
// Softmax denominator accumulated with a ones-B WMMA (no lane reduction).
// ---------------------------------------------------------------------------
__global__ void __launch_bounds__(128) flash_attn_wmma(
    const __bf16* __restrict__ Q, const __bf16* __restrict__ K,
    const __bf16* __restrict__ V, const int* __restrict__ gmask,
    const int* __restrict__ amask, __bf16* __restrict__ Ctx) {
  __shared__ __bf16 Ks[32][HDIM];   // [key][dim], 128B rows
  __shared__ __bf16 Vt[HDIM][40];   // [dim][key], 80B rows: 16B-aligned, conflict-free
  __shared__ __bf16 Ps[4][16][32];  // per-wave P tile for D->A re-layout

  const int lane = threadIdx.x & 31;
  const int wid  = threadIdx.x >> 5;
  const int half = lane >> 4;
  const int l16  = lane & 15;
  const int h = blockIdx.y;
  const int b = blockIdx.z;
  const int q0 = blockIdx.x * 64 + wid * 16;

  const size_t rowBase = (size_t)b * SEQ;
  const __bf16* Qb = Q + rowBase * HIDDEN + h * HDIM;
  const __bf16* Kb = K + rowBase * HIDDEN + h * HDIM;
  const __bf16* Vb = V + rowBase * HIDDEN + h * HDIM;

  // Q tile as two resident A-fragments (k = 0..31, 32..63)
  const __bf16* qrow = Qb + (size_t)(q0 + l16) * HIDDEN;
  const v16bf aq0 = ldA(qrow + 16 * half);
  const v16bf aq1 = ldA(qrow + 32 + 16 * half);

  // constant ones B-fragment for the row-sum WMMA
  v16bf onesB;
#pragma unroll
  for (int i = 0; i < 16; ++i) onesB[i] = (__bf16)1.0f;

  int  qi[8];
  bool grow[8];
  float mrun[8];
#pragma unroll
  for (int r = 0; r < 8; ++r) {
    qi[r]   = q0 + r + 8 * half;            // C/D layout: row = r + 8*half
    grow[r] = gmask[rowBase + qi[r]] != 0;
    mrun[r] = -3.0e38f;
  }
  v8f acc[4] = {};                          // 16 x 64 output accumulator (f32)
  v8f accL = {};                            // 16 x 16 row-sum accumulator

  const int key   = threadIdx.x >> 2;       // staging: 4 threads / key row
  const int dbase = (threadIdx.x & 3) * 16;

  for (int kc = 0; kc < SEQ / 32; ++kc) {
    const int k0 = kc * 32;
    __syncthreads();                        // previous chunk fully consumed
    {
      const __bf16* ks = Kb + (size_t)(k0 + key) * HIDDEN + dbase;
      const __bf16* vs = Vb + (size_t)(k0 + key) * HIDDEN + dbase;
#if HAVE_ASYNC_LDS
      ASYNC_B128(ks, &Ks[key][dbase], 0);   // 32B per thread, VGPR-bypassing
      ASYNC_B128(ks, &Ks[key][dbase], 16);
#else
#pragma unroll
      for (int i = 0; i < 16; ++i) Ks[key][dbase + i] = ks[i];
#endif
#pragma unroll
      for (int i = 0; i < 16; ++i) Vt[dbase + i][key] = vs[i];
      if (k0 + 32 < SEQ) {
        __builtin_prefetch(ks + 32 * HIDDEN, 0, 0);
        __builtin_prefetch(vs + 32 * HIDDEN, 0, 0);
      }
    }
    ASYNC_WAIT();
    __syncthreads();

    // S = Q * K^T  (two 16-key column tiles, K-dim = 64 -> 4 WMMA)
    v8f s0 = {}, s1 = {};
    s0 = wmma_bf16(aq0, ldB(&Ks[l16][16 * half]), s0);
    s0 = wmma_bf16(aq1, ldB(&Ks[l16][32 + 16 * half]), s0);
    s1 = wmma_bf16(aq0, ldB(&Ks[16 + l16][16 * half]), s1);
    s1 = wmma_bf16(aq1, ldB(&Ks[16 + l16][32 + 16 * half]), s1);

    const int  kj0 = k0 + l16, kj1 = kj0 + 16;
    const bool ok0 = amask[rowBase + kj0] != 0;
    const bool ok1 = amask[rowBase + kj1] != 0;
    const bool gc0 = gmask[rowBase + kj0] != 0;
    const bool gc1 = gmask[rowBase + kj1] != 0;

#pragma unroll
    for (int r = 0; r < 8; ++r) {
      const int d0 = qi[r] - kj0, d1 = qi[r] - kj1;
      const bool w0 = (d0 <= WIN) && (d0 >= -WIN);
      const bool w1 = (d1 <= WIN) && (d1 >= -WIN);
      const float sc0 = ((w0 || grow[r] || gc0) && ok0) ? s0[r] * 0.125f : NEGV;
      const float sc1 = ((w1 || grow[r] || gc1) && ok1) ? s1[r] * 0.125f : NEGV;

      const float mt   = rowmax16(fmaxf(sc0, sc1));     // 4x ds_swizzle
      const float mnew = fmaxf(mrun[r], mt);
      const float alpha = __expf(mrun[r] - mnew);
      mrun[r] = mnew;
      const float p0 = __expf(sc0 - mnew);
      const float p1 = __expf(sc1 - mnew);

      accL[r] *= alpha;                                  // rescale denominator
#pragma unroll
      for (int nt = 0; nt < 4; ++nt) acc[nt][r] *= alpha;

      // stage P (D-layout) into LDS for A-layout reload (intra-wave, in-order)
      Ps[wid][r + 8 * half][l16]      = (__bf16)p0;
      Ps[wid][r + 8 * half][l16 + 16] = (__bf16)p1;
    }

    // O += P(16x32) * V(32x64); L += P * ones  -> 5 WMMA
    const v16bf pa = ldA(&Ps[wid][l16][16 * half]);
    accL = wmma_bf16(pa, onesB, accL);
#pragma unroll
    for (int nt = 0; nt < 4; ++nt)
      acc[nt] = wmma_bf16(pa, ldB(&Vt[nt * 16 + l16][16 * half]), acc[nt]);
  }

#pragma unroll
  for (int r = 0; r < 8; ++r) {
    const float l   = accL[r];
    const float inv = (l > 0.0f) ? 1.0f / l : 0.0f;
    const int row = q0 + r + 8 * half;
    __bf16* crow = Ctx + (rowBase + row) * HIDDEN + h * HDIM;
#pragma unroll
    for (int nt = 0; nt < 4; ++nt)
      crow[nt * 16 + l16] = (__bf16)(acc[nt][r] * inv);
  }
}

// ---------------------------------------------------------------------------
extern "C" void kernel_launch(void* const* d_in, const int* in_sizes, int n_in,
                              void* d_out, int out_size, void* d_ws, size_t ws_size,
                              hipStream_t stream) {
  const float* hidden = (const float*)d_in[0];
  const int*   amask  = (const int*)d_in[1];
  const int*   gmask  = (const int*)d_in[2];
  const float* Wq = (const float*)d_in[3];  const float* bq = (const float*)d_in[4];
  const float* Wk = (const float*)d_in[5];  const float* bk = (const float*)d_in[6];
  const float* Wv = (const float*)d_in[7];  const float* bv = (const float*)d_in[8];
  const float* Wo = (const float*)d_in[9];  const float* bo = (const float*)d_in[10];
  float* out = (float*)d_out;

  const int Bb = in_sizes[0] / (SEQ * HIDDEN);   // batch (=2)
  const int M  = Bb * SEQ;                       // 4096 rows
  const int WN = HIDDEN * HIDDEN;                // 589824

  // workspace carve-up (all bf16), 256B aligned
  char* ws = (char*)d_ws;
  size_t off = 0;
  auto carve = [&](size_t bytes) {
    size_t o = off;
    off = (off + bytes + 255) & ~(size_t)255;
    return (void*)(ws + o);
  };
  __bf16* Xbf = (__bf16*)carve((size_t)M * HIDDEN * 2);
  __bf16* Wqb = (__bf16*)carve((size_t)WN * 2);
  __bf16* Wkb = (__bf16*)carve((size_t)WN * 2);
  __bf16* Wvb = (__bf16*)carve((size_t)WN * 2);
  __bf16* Wob = (__bf16*)carve((size_t)WN * 2);
  __bf16* Qbf = (__bf16*)carve((size_t)M * HIDDEN * 2);
  __bf16* Kbf = (__bf16*)carve((size_t)M * HIDDEN * 2);
  __bf16* Vbf = (__bf16*)carve((size_t)M * HIDDEN * 2);
  __bf16* Ctx = Xbf;  // X is dead after the QKV projections; reuse its slot
  (void)ws_size; (void)n_in; (void)out_size;

  // 1) one-time fp32 -> bf16 casts
  const int nx = M * HIDDEN;
  cvt_f32_to_bf16<<<(nx + 255) / 256, 256, 0, stream>>>(hidden, Xbf, nx);
  cvt_f32_to_bf16<<<(WN + 255) / 256, 256, 0, stream>>>(Wq, Wqb, WN);
  cvt_f32_to_bf16<<<(WN + 255) / 256, 256, 0, stream>>>(Wk, Wkb, WN);
  cvt_f32_to_bf16<<<(WN + 255) / 256, 256, 0, stream>>>(Wv, Wvb, WN);
  cvt_f32_to_bf16<<<(WN + 255) / 256, 256, 0, stream>>>(Wo, Wob, WN);

  // 2) Q/K/V projections on WMMA
  const int tiles   = (M >> 5) * (HIDDEN >> 5);
  const int gblocks = (tiles + 7) / 8;
  gemm_bias_wmma<true><<<gblocks, 256, 0, stream>>>(Xbf, Wqb, bq, Qbf, M, HIDDEN, HIDDEN);
  gemm_bias_wmma<true><<<gblocks, 256, 0, stream>>>(Xbf, Wkb, bk, Kbf, M, HIDDEN, HIDDEN);
  gemm_bias_wmma<true><<<gblocks, 256, 0, stream>>>(Xbf, Wvb, bv, Vbf, M, HIDDEN, HIDDEN);

  // 3) masked flash attention (window | global-row | global-col, key pad mask)
  dim3 agrid(SEQ / 64, NHEADS, Bb);
  flash_attn_wmma<<<agrid, 128, 0, stream>>>(Qbf, Kbf, Vbf, gmask, amask, Ctx);

  // 4) output projection -> fp32 d_out
  gemm_bias_wmma<false><<<gblocks, 256, 0, stream>>>(Ctx, Wob, bo, out, M, HIDDEN, HIDDEN);
}
